// PositionalEncoding_39041252721255
// MI455X (gfx1250) — matrix operand order repaired and checked by reading, not compiled, not run on hardware
//
#include <hip/hip_runtime.h>
#include <stdint.h>

typedef __attribute__((ext_vector_type(16))) _Float16 v16h;
typedef __attribute__((ext_vector_type(8)))  float    v8f;

#define S_LEN   2048
#define DM      512
#define POS_PER_BLK 8

// ---------------------------------------------------------------------------
// Kernel 1: per-row rank = cumsum(mask)-1 via WMMA-accelerated block scan.
// grid = B, block = 256. Thread t owns mask bytes [8t, 8t+8) of its row.
// Wave 0 scans the 256 thread-sums with two v_wmma_f32_16x16x32_f16:
//   P = V x U          (U[k,n] = k<=n, k<16)   -> within-group-of-16 prefix
//   S = Lstrict x T + P (L[m,k] = k<m, B[k,n]=T[k]) -> full 256 inclusive scan
// All values are small integers: exact in f16 inputs / f32 accumulate.
// rank out: masked -> 0-based rank, unmasked -> -1 sentinel.
// ---------------------------------------------------------------------------
__global__ void __launch_bounds__(256)
pe_rank_scan(const unsigned char* __restrict__ masks, int* __restrict__ rank)
{
    __shared__ float sdata[256];   // thread sums as V[16][16]
    __shared__ float tdata[16];    // group totals T[i] = P[i][15]
    __shared__ float scanS[256];   // inclusive scan of thread sums

    const int t    = threadIdx.x;
    const int row  = blockIdx.x;
    const int base = row * S_LEN + t * 8;

    // 8 mask bytes (each 0/1, jax bool layout)
    uint2 mw = *(const uint2*)(masks + base);
    unsigned char mb[8];
    int s = 0;
#pragma unroll
    for (int i = 0; i < 4; ++i) { mb[i]     = (mw.x >> (8 * i)) & 1; s += mb[i];     }
#pragma unroll
    for (int i = 0; i < 4; ++i) { mb[4 + i] = (mw.y >> (8 * i)) & 1; s += mb[4 + i]; }

    sdata[t] = (float)s;
    __syncthreads();

    if (t < 32) {                      // wave 0 only; EXEC all-ones in-wave
        const int lane = t;
        const int m    = lane & 15;    // A row / B col / D col for this lane
        const int hi   = lane >> 4;

        // --- WMMA #1 : P = V x U ------------------------------------------
        v16h a1, b1;
#pragma unroll
        for (int r = 0; r < 8; ++r) {
            // 16-bit A 16x32 layout (ISA 7.12.2): lane<16 K={2r | 2r+8},
            // lane>=16 K={2r+8 | 2r+16}
            int k0 = hi ? ((r < 4) ? 2 * r + 8 : 2 * r + 16)
                        : ((r < 4) ? 2 * r     : 2 * r + 8);
            a1[2 * r]     = (k0     < 16) ? (_Float16)sdata[16 * m + k0]     : (_Float16)0.f;
            a1[2 * r + 1] = (k0 + 1 < 16) ? (_Float16)sdata[16 * m + k0 + 1] : (_Float16)0.f;
            // 16-bit B 32x16 layout: same K order as A
            int kb = k0;
            b1[2 * r]     = ((kb     < 16) && (kb     <= m)) ? (_Float16)1.f : (_Float16)0.f;
            b1[2 * r + 1] = ((kb + 1 < 16) && (kb + 1 <= m)) ? (_Float16)1.f : (_Float16)0.f;
        }
        v8f czero = {};
        v8f d1 = __builtin_amdgcn_wmma_f32_16x16x32_f16(false, a1, false, b1,
                                                        (short)0, czero, false, false);

        // stash group totals T[i] = P[i][15]  (lane 15 -> i=0..7, lane 31 -> 8..15)
        if (m == 15) {
#pragma unroll
            for (int v = 0; v < 8; ++v) tdata[hi ? v + 8 : v] = d1[v];
        }
        asm volatile("s_wait_dscnt 0" ::: "memory");   // in-wave LDS RAW fence

        // --- WMMA #2 : S = Lstrict x Trep + P -----------------------------
        v16h a2, b2;
#pragma unroll
        for (int r = 0; r < 8; ++r) {
            int k0 = hi ? ((r < 4) ? 2 * r + 8 : 2 * r + 16)
                        : ((r < 4) ? 2 * r     : 2 * r + 8);
            a2[2 * r]     = ((k0     < 16) && (k0     < m)) ? (_Float16)1.f : (_Float16)0.f;
            a2[2 * r + 1] = ((k0 + 1 < 16) && (k0 + 1 < m)) ? (_Float16)1.f : (_Float16)0.f;
            int kb = k0;
            b2[2 * r]     = (kb     < 16) ? (_Float16)tdata[kb]     : (_Float16)0.f;
            b2[2 * r + 1] = (kb + 1 < 16) ? (_Float16)tdata[kb + 1] : (_Float16)0.f;
        }
        v8f d2 = __builtin_amdgcn_wmma_f32_16x16x32_f16(false, a2, false, b2,
                                                        (short)0, d1, false, false);

        // f32 D 16x16 layout: VGPR v, lanes0-15 -> M=v ; lanes16-31 -> M=v+8
#pragma unroll
        for (int v = 0; v < 8; ++v)
            scanS[16 * (hi ? v + 8 : v) + m] = d2[v];
    }
    __syncthreads();

    int run = (int)scanS[t] - s;       // exclusive base for this thread
#pragma unroll
    for (int i = 0; i < 8; ++i) {
        run += mb[i];
        rank[base + i] = mb[i] ? (run - 1) : -1;
    }
}

// ---------------------------------------------------------------------------
// Kernel 2: out = seqs + (rank>=0 ? pe[rank] : 0).
// Block covers 8 positions (8 x 512 f32). The dependent pe-row gather is
// staged to LDS with global_load_async_to_lds_b128 and overlapped with the
// independent b128 streaming loads of the seq tile (ASYNCcnt path).
// ---------------------------------------------------------------------------
__global__ void __launch_bounds__(256)
pe_add(const float* __restrict__ seqs, const float* __restrict__ pe,
       const int* __restrict__ rank, float* __restrict__ out)
{
    __shared__ __align__(16) float pebuf[POS_PER_BLK * DM];   // 16 KB
    __shared__ int rbuf[POS_PER_BLK];

    const int t    = threadIdx.x;
    const int pos0 = blockIdx.x * POS_PER_BLK;                // flat b*S + s

    if (t < POS_PER_BLK) rbuf[t] = rank[pos0 + t];
    __syncthreads();

    // issue 4 x b128 async copies per thread: 8 rows x 2 KB -> LDS
#pragma unroll
    for (int i = 0; i < 4; ++i) {
        int c   = i * 256 + t;        // chunk 0..1023, lanes stay coalesced
        int pos = c >> 7;             // 128 x 16B chunks per position
        int cj  = c & 127;
        int r   = rbuf[pos];
        int rr  = (r < 0) ? 0 : r;
        const float* src = pe + (size_t)rr * DM + cj * 4;
        unsigned lds = (unsigned)(uintptr_t)&pebuf[pos * DM + cj * 4];
        asm volatile("global_load_async_to_lds_b128 %0, %1, off"
                     :: "v"(lds), "v"(src) : "memory");
    }

    // overlap: stream the independent seq tile while the async copies fly
    const float4* sq = (const float4*)(seqs + (size_t)pos0 * DM);
    float4 sv[4];
#pragma unroll
    for (int i = 0; i < 4; ++i) sv[i] = sq[i * 256 + t];

    asm volatile("s_wait_asynccnt 0" ::: "memory");
    __syncthreads();

    float4* op = (float4*)(out + (size_t)pos0 * DM);
#pragma unroll
    for (int i = 0; i < 4; ++i) {
        int e   = i * 256 + t;
        int pos = e >> 7;
        int ej  = e & 127;
        float4 v = sv[i];
        if (rbuf[pos] >= 0) {
            const float4 p = *(const float4*)&pebuf[pos * DM + ej * 4];
            v.x += p.x; v.y += p.y; v.z += p.z; v.w += p.w;
        }
        op[e] = v;
    }
}

// ---------------------------------------------------------------------------
extern "C" void kernel_launch(void* const* d_in, const int* in_sizes, int n_in,
                              void* d_out, int out_size, void* d_ws, size_t ws_size,
                              hipStream_t stream)
{
    const float*         seqs  = (const float*)d_in[0];
    const unsigned char* masks = (const unsigned char*)d_in[1];  // jax bool = 1 byte
    const float*         pe    = (const float*)d_in[2];
    float*               out   = (float*)d_out;
    int*                 rank  = (int*)d_ws;                     // B*S ints (256 KB)

    const int B = in_sizes[0] / (S_LEN * DM);                    // 32

    pe_rank_scan<<<B, 256, 0, stream>>>(masks, rank);
    pe_add<<<(B * S_LEN) / POS_PER_BLK, 256, 0, stream>>>(seqs, pe, rank, out);
}
